// CrossAttention_16363825398343
// MI455X (gfx1250) — compile-verified, matrix operand order checked by reading, not run on hardware
//
#include <hip/hip_runtime.h>

typedef __attribute__((ext_vector_type(16))) _Float16 v16h;
typedef __attribute__((ext_vector_type(8)))  float    v8f;

#define DD   512
#define FF   2048
#define TKV  4096
#define TQ   512
#define BATCH 8
#define CHUNK 128

// ---------------------------------------------------------------------------
// Fragment loader: 16-bit A (16xK) / B (Kx16) fragment.
// lane L: row/col = rc + (L&15); 16 contiguous halves at k = kbase + 16*(L>>4)
// Works for global or LDS pointers (generic address space).
// ---------------------------------------------------------------------------
__device__ __forceinline__ v16h load_frag(const _Float16* base, int stride,
                                          int rc, int kbase, int lane) {
  const _Float16* p = base + (size_t)(rc + (lane & 15)) * stride
                           + kbase + ((lane >> 4) << 4);
  return *(const v16h*)p;
}

__device__ __forceinline__ v8f wmma_f16(v16h a, v16h b, v8f c) {
  return __builtin_amdgcn_wmma_f32_16x16x32_f16(false, a, false, b,
                                                (short)0, c, false, false);
}

// ---------------------------------------------------------------------------
// LayerNorm over last dim (512), optional f32 and f16 outputs.
// One 256-thread block per row.
// ---------------------------------------------------------------------------
__global__ __launch_bounds__(256)
void ln_kernel(const float* __restrict__ x, const float* __restrict__ g,
               const float* __restrict__ b, float* __restrict__ y32,
               _Float16* __restrict__ y16) {
  __shared__ float red[8];
  __shared__ float bval;
  size_t row = blockIdx.x;
  const float* xr = x + row * DD;
  int t = threadIdx.x;
  float a0 = xr[t], a1 = xr[t + 256];

  float s = a0 + a1;
  for (int off = 16; off; off >>= 1) s += __shfl_down(s, off, 32);
  if ((t & 31) == 0) red[t >> 5] = s;
  __syncthreads();
  if (t == 0) { float tt = 0.f; for (int i = 0; i < 8; ++i) tt += red[i];
                bval = tt * (1.f / (float)DD); }
  __syncthreads();
  float mu = bval;

  float d0 = a0 - mu, d1 = a1 - mu;
  float vs = d0 * d0 + d1 * d1;
  for (int off = 16; off; off >>= 1) vs += __shfl_down(vs, off, 32);
  if ((t & 31) == 0) red[t >> 5] = vs;
  __syncthreads();
  if (t == 0) { float tt = 0.f; for (int i = 0; i < 8; ++i) tt += red[i];
                bval = tt * (1.f / (float)DD); }
  __syncthreads();
  float rstd = rsqrtf(bval + 1e-6f);

  float o0 = d0 * rstd * g[t]       + b[t];
  float o1 = d1 * rstd * g[t + 256] + b[t + 256];
  if (y32) { float* yr = y32 + row * DD; yr[t] = o0; yr[t + 256] = o1; }
  if (y16) { _Float16* yr = y16 + row * DD;
             yr[t] = (_Float16)o0; yr[t + 256] = (_Float16)o1; }
}

// f32 -> f16 convert
__global__ __launch_bounds__(256)
void cvt_kernel(const float* __restrict__ x, _Float16* __restrict__ y, int n) {
  int i = blockIdx.x * 256 + threadIdx.x;
  if (i < n) y[i] = (_Float16)x[i];
}

// W[R,C] f32 -> Wt[C,R] f16 (so B-fragments read contiguous K per lane)
__global__ __launch_bounds__(256)
void cvt_transpose_kernel(const float* __restrict__ W, _Float16* __restrict__ Wt,
                          int R, int C) {
  int i = blockIdx.x * 256 + threadIdx.x;
  if (i < R * C) {
    int r = i / C, c = i % C;
    Wt[(size_t)c * R + r] = (_Float16)W[i];
  }
}

// ---------------------------------------------------------------------------
// WMMA GEMM: C[M,N] = A[M,K] * Bt[N,K]^T + bias
// Block tile 64x128 (8 waves, each 32x32 = 2x2 WMMA tiles), K-step 32.
// Optional outputs: f16 row-major, f32 row-major (+residual), f16 transposed
// per-batch (CT[(m/TB)*N + n][TB], used to build V^T).
// ---------------------------------------------------------------------------
__global__ __launch_bounds__(256)
void gemm_wmma_kernel(const _Float16* __restrict__ A,
                      const _Float16* __restrict__ Bt,
                      const float* __restrict__ bias,
                      _Float16* __restrict__ C16,
                      float* __restrict__ C32,
                      const float* __restrict__ resid,
                      _Float16* __restrict__ CT16,
                      int M, int N, int K, int TB) {
  int lane = threadIdx.x & 31;
  int wave = threadIdx.x >> 5;       // 0..7
  int wm = wave >> 2;                // 0..1
  int wn = wave & 3;                 // 0..3
  int m0 = blockIdx.y * 64  + wm * 32;
  int n0 = blockIdx.x * 128 + wn * 32;

  v8f acc[2][2] = {};
  for (int kb = 0; kb < K; kb += 32) {
    v16h a0 = load_frag(A,  K, m0,      kb, lane);
    v16h a1 = load_frag(A,  K, m0 + 16, kb, lane);
    v16h b0 = load_frag(Bt, K, n0,      kb, lane);
    v16h b1 = load_frag(Bt, K, n0 + 16, kb, lane);
    acc[0][0] = wmma_f16(a0, b0, acc[0][0]);
    acc[0][1] = wmma_f16(a0, b1, acc[0][1]);
    acc[1][0] = wmma_f16(a1, b0, acc[1][0]);
    acc[1][1] = wmma_f16(a1, b1, acc[1][1]);
  }

  for (int i = 0; i < 2; ++i)
    for (int j = 0; j < 2; ++j) {
      int col = n0 + j * 16 + (lane & 15);
      float bv = bias ? bias[col] : 0.f;
      for (int r = 0; r < 8; ++r) {
        int row = m0 + i * 16 + r + ((lane >> 4) << 3);
        float val = acc[i][j][r] + bv;
        size_t idx = (size_t)row * N + col;
        if (C32) C32[idx] = resid ? val + resid[idx] : val;
        if (C16) C16[idx] = (_Float16)val;
        if (CT16) {
          int bb = row / TB, tt = row % TB;
          CT16[((size_t)bb * N + col) * TB + tt] = (_Float16)val;
        }
      }
    }
}

// ---------------------------------------------------------------------------
// Flash-style attention. One block per (batch, 16 q-rows). 8 waves.
// scores = (q.k^T + qpos.kpos^T)/sqrt(D+1e-7); online softmax over Tkv in
// chunks of 128; O += P.V with V read from V^T; residual with raw query.
// q/qpos tiles staged into LDS via CDNA5 async-to-LDS DMA (ASYNCcnt).
// ---------------------------------------------------------------------------
__global__ __launch_bounds__(256)
void attn_kernel(const _Float16* __restrict__ q16,   // [B*Tq, D]
                 const _Float16* __restrict__ qp16,  // [B*Tq, D]
                 const _Float16* __restrict__ k16,   // [B*Tkv, D]
                 const _Float16* __restrict__ kp16,  // [B*Tkv, D]
                 const _Float16* __restrict__ vT16,  // [B, D, Tkv]
                 const float*    __restrict__ query, // [B*Tq, D]
                 float*          __restrict__ out) { // [B*Tq, D]
  __shared__ __align__(32) _Float16 sQ[16 * DD];
  __shared__ __align__(32) _Float16 sQP[16 * DD];
  __shared__ float    sS[16 * CHUNK];
  __shared__ __align__(32) _Float16 sP[16 * CHUNK];
  __shared__ float sM[16], sL[16], sScale[16];

  int b   = blockIdx.x >> 5;          // Tq/16 = 32 q-tiles per batch
  int qt  = blockIdx.x & 31;
  int row0 = b * TQ + qt * 16;        // global q-row
  int lane = threadIdx.x & 31;
  int wave = threadIdx.x >> 5;
  const float scale = 0.04419417382f; // 1/sqrt(512 + 1e-7)

  { // stage q and qpos tiles into LDS with async-to-LDS b128 DMA
    const _Float16* g1 = q16  + (size_t)row0 * DD;
    const _Float16* g2 = qp16 + (size_t)row0 * DD;
    unsigned lq0 = (unsigned)(size_t)(void*)sQ;
    unsigned lp0 = (unsigned)(size_t)(void*)sQP;
    for (int i = threadIdx.x; i < 16 * DD / 8; i += 256) {
      const _Float16* p1 = g1 + i * 8;       // 8 halves = 16B per op
      const _Float16* p2 = g2 + i * 8;
      unsigned l1 = lq0 + (unsigned)i * 16;  // LDS byte offset (wave-relative)
      unsigned l2 = lp0 + (unsigned)i * 16;
      asm volatile("global_load_async_to_lds_b128 %0, %1, off"
                   :: "v"(l1), "v"(p1) : "memory");
      asm volatile("global_load_async_to_lds_b128 %0, %1, off"
                   :: "v"(l2), "v"(p2) : "memory");
    }
    asm volatile("s_wait_asynccnt 0x0" ::: "memory");
  }
  if (threadIdx.x < 16) { sM[threadIdx.x] = -1e30f; sL[threadIdx.x] = 0.f; }
  __syncthreads();

  v8f o[4] = {};  // this wave's 64 output cols: [wave*64 + t*16, +16)

  for (int kc = 0; kc < TKV / CHUNK; ++kc) {
    int kv0 = kc * CHUNK;
    // --- scores: wave handles kv cols [kv0 + wave*16, +16), K = D = 512
    const _Float16* kb_  = k16  + ((size_t)b * TKV + kv0 + wave * 16) * DD;
    const _Float16* kpb_ = kp16 + ((size_t)b * TKV + kv0 + wave * 16) * DD;
    v8f s = {};
    for (int kk = 0; kk < DD; kk += 32) {
      v16h aq = load_frag(sQ,  DD, 0, kk, lane);
      v16h bk = load_frag(kb_, DD, 0, kk, lane);
      s = wmma_f16(aq, bk, s);
      v16h ap = load_frag(sQP, DD, 0, kk, lane);
      v16h bp = load_frag(kpb_, DD, 0, kk, lane);
      s = wmma_f16(ap, bp, s);
    }
    for (int r = 0; r < 8; ++r) {
      int row = r + ((lane >> 4) << 3);
      sS[row * CHUNK + wave * 16 + (lane & 15)] = s[r] * scale;
    }
    __syncthreads();

    // --- online softmax update, parallel over all 256 threads:
    // 16 threads per q-row (contiguous lanes), 8 columns each,
    // butterfly reductions within the 16-lane groups.
    {
      int row = threadIdx.x >> 4;
      int j   = threadIdx.x & 15;
      float vals[8];
      float mx = -1e30f;
      for (int c = 0; c < 8; ++c) {
        vals[c] = sS[row * CHUNK + j + c * 16];
        mx = fmaxf(mx, vals[c]);
      }
      for (int m = 8; m >= 1; m >>= 1) mx = fmaxf(mx, __shfl_xor(mx, m, 32));
      float m_old = sM[row];
      float m_new = fmaxf(m_old, mx);
      float sum = 0.f;
      for (int c = 0; c < 8; ++c) {
        float p = __expf(vals[c] - m_new);
        sum += p;
        sP[row * CHUNK + j + c * 16] = (_Float16)p;
      }
      for (int m = 8; m >= 1; m >>= 1) sum += __shfl_xor(sum, m, 32);
      if (j == 0) {
        float sc = __expf(m_old - m_new);
        sM[row] = m_new;
        sL[row] = sL[row] * sc + sum;
        sScale[row] = sc;
      }
    }
    __syncthreads();

    // --- rescale accumulators, then O += P @ V_chunk
    for (int t = 0; t < 4; ++t)
      for (int r = 0; r < 8; ++r)
        o[t][r] *= sScale[r + ((lane >> 4) << 3)];

    const _Float16* vb = vT16 + ((size_t)b * DD + wave * 64) * TKV + kv0;
    for (int kk = 0; kk < CHUNK; kk += 32) {
      v16h ap = load_frag(sP, CHUNK, 0, kk, lane);
      for (int t = 0; t < 4; ++t) {
        v16h bv = load_frag(vb + (size_t)t * 16 * TKV, TKV, 0, kk, lane);
        o[t] = wmma_f16(ap, bv, o[t]);
      }
    }
    __syncthreads();
  }

  // --- finalize: O/l + raw-query residual
  for (int t = 0; t < 4; ++t) {
    int col = wave * 64 + t * 16 + (lane & 15);
    for (int r = 0; r < 8; ++r) {
      int row = r + ((lane >> 4) << 3);
      float inv = 1.0f / sL[row];
      size_t idx = (size_t)(row0 + row) * DD + col;
      out[idx] = o[t][r] * inv + query[idx];
    }
  }
}

// ---------------------------------------------------------------------------
// Host-side orchestration
// ---------------------------------------------------------------------------
extern "C" void kernel_launch(void* const* d_in, const int* in_sizes, int n_in,
                              void* d_out, int out_size, void* d_ws, size_t ws_size,
                              hipStream_t stream) {
  const float* query     = (const float*)d_in[0];
  const float* key_value = (const float*)d_in[1];
  const float* query_pos = (const float*)d_in[2];
  const float* key_pos   = (const float*)d_in[3];
  const float* q_gamma   = (const float*)d_in[4];
  const float* q_beta    = (const float*)d_in[5];
  const float* kv_gamma  = (const float*)d_in[6];
  const float* kv_beta   = (const float*)d_in[7];
  const float* Wq = (const float*)d_in[8];   const float* bq = (const float*)d_in[9];
  const float* Wk = (const float*)d_in[10];  const float* bk = (const float*)d_in[11];
  const float* Wv = (const float*)d_in[12];  const float* bv = (const float*)d_in[13];
  const float* ff_gamma = (const float*)d_in[14];
  const float* ff_beta  = (const float*)d_in[15];
  const float* W_inner  = (const float*)d_in[16];
  const float* b_inner  = (const float*)d_in[17];
  const float* W_proj   = (const float*)d_in[18];
  const float* b_proj   = (const float*)d_in[19];

  const int MQ  = BATCH * TQ;    // 4096
  const int MKV = BATCH * TKV;   // 32768

  size_t off = 0;
  auto alloc = [&](size_t bytes) -> void* {
    void* p = (char*)d_ws + off;
    off = (off + bytes + 255) & ~(size_t)255;
    return p;
  };
  _Float16* qn16  = (_Float16*)alloc((size_t)MQ  * DD * 2);
  _Float16* kvn16 = (_Float16*)alloc((size_t)MKV * DD * 2);   // reused as inner16 later
  _Float16* qp16  = (_Float16*)alloc((size_t)MQ  * DD * 2);   // reused as xn16 later
  _Float16* kp16  = (_Float16*)alloc((size_t)MKV * DD * 2);   // reused as xn32 later
  _Float16* WqT   = (_Float16*)alloc((size_t)DD * DD * 2);
  _Float16* WkT   = (_Float16*)alloc((size_t)DD * DD * 2);
  _Float16* WvT   = (_Float16*)alloc((size_t)DD * DD * 2);
  _Float16* WinT  = (_Float16*)alloc((size_t)FF * DD * 2);    // [N=2048][K=512]
  _Float16* WprT  = (_Float16*)alloc((size_t)DD * FF * 2);    // [N=512][K=2048]
  _Float16* q16   = (_Float16*)alloc((size_t)MQ  * DD * 2);
  _Float16* k16   = (_Float16*)alloc((size_t)MKV * DD * 2);
  _Float16* vT16  = (_Float16*)alloc((size_t)BATCH * DD * TKV * 2);
  float*    aout  = (float*)   alloc((size_t)MQ * DD * 4);
  // aliases (lifetimes are disjoint, stream-ordered):
  _Float16* inner16 = kvn16;                  // [4096,2048] f16 = 16 MiB <= 32 MiB
  _Float16* xn16    = qp16;                   // [4096,512] f16
  float*    xn32    = (float*)kp16;           // [4096,512] f32 = 8 MiB <= 32 MiB

  // 1) weight converts (transposed to [N][K])
  cvt_transpose_kernel<<<(DD*DD)/256, 256, 0, stream>>>(Wq, WqT, DD, DD);
  cvt_transpose_kernel<<<(DD*DD)/256, 256, 0, stream>>>(Wk, WkT, DD, DD);
  cvt_transpose_kernel<<<(DD*DD)/256, 256, 0, stream>>>(Wv, WvT, DD, DD);
  cvt_transpose_kernel<<<(DD*FF)/256, 256, 0, stream>>>(W_inner, WinT, DD, FF);
  cvt_transpose_kernel<<<(FF*DD)/256, 256, 0, stream>>>(W_proj,  WprT, FF, DD);

  // 2) positional tensors -> f16
  cvt_kernel<<<(MQ*DD)/256,  256, 0, stream>>>(query_pos, qp16, MQ*DD);
  cvt_kernel<<<(MKV*DD)/256, 256, 0, stream>>>(key_pos,   kp16, MKV*DD);

  // 3) LayerNorms -> f16
  ln_kernel<<<MQ,  256, 0, stream>>>(query,     q_gamma,  q_beta,  nullptr, qn16);
  ln_kernel<<<MKV, 256, 0, stream>>>(key_value, kv_gamma, kv_beta, nullptr, kvn16);

  // 4) Q/K/V projections (WMMA)
  gemm_wmma_kernel<<<dim3(DD/128, MQ/64),  256, 0, stream>>>(
      qn16,  WqT, bq, q16,  nullptr, nullptr, nullptr, MQ,  DD, DD, TKV);
  gemm_wmma_kernel<<<dim3(DD/128, MKV/64), 256, 0, stream>>>(
      kvn16, WkT, bk, k16,  nullptr, nullptr, nullptr, MKV, DD, DD, TKV);
  gemm_wmma_kernel<<<dim3(DD/128, MKV/64), 256, 0, stream>>>(
      kvn16, WvT, bv, nullptr, nullptr, nullptr, vT16, MKV, DD, DD, TKV);

  // 5) attention (flash-style, WMMA scores + WMMA P.V)
  attn_kernel<<<BATCH * (TQ/16), 256, 0, stream>>>(
      q16, qp16, k16, kp16, vT16, query, aout);

  // 6) FF: xn = LN(out); d_out = xn + (xn@W_inner + b_inner)@W_proj + b_proj
  ln_kernel<<<MQ, 256, 0, stream>>>(aout, ff_gamma, ff_beta, xn32, xn16);
  gemm_wmma_kernel<<<dim3(FF/128, MQ/64), 256, 0, stream>>>(
      xn16, WinT, b_inner, inner16, nullptr, nullptr, nullptr, MQ, FF, DD, TKV);
  gemm_wmma_kernel<<<dim3(DD/128, MQ/64), 256, 0, stream>>>(
      inner16, WprT, b_proj, nullptr, (float*)d_out, xn32, nullptr, MQ, DD, FF, TKV);
}
